// GVPDiff_80427557585234
// MI455X (gfx1250) — matrix operand order repaired and checked by reading, not compiled, not run on hardware
//
#include <hip/hip_runtime.h>
#include <math.h>

// ---------------------------------------------------------------------------
// GVP-GNN diffusion model for MI455X (gfx1250), wave32 + WMMA bf16.
//
// All dense linear layers run through one generic WMMA GEMM kernel
// (v_wmma_f32_16x16x32_bf16, fp32 accumulate) whose A operand is assembled in
// LDS from up to 4 "segments" (optionally row-gathered via an index array, or
// broadcast via rowStride==0). This fuses the per-edge gather+concat
// (s[src] || es || s[dst] || |vh|) into the GEMM so the 300MB message
// tensors are never materialized -> HBM-roofline friendly.
//
// Gather codegen: per-row 32-bit offsets (rowIdx*stride, -1 = inactive) are
// resolved once per block into LDS; segment base pointers stay in registers so
// the hot-loop A loads lower to global_load (not flat_load). The per-column
// segment search is hoisted out of the row loop (column is fixed per thread).
// NOTE: all rowIdx*stride products in this model are < 2^31 (max 240000*544).
//
// Param-leaf assumption: the harness appends params flattened recursively in
// dict *insertion* order after the 8 main inputs:
//   Wv{ln{g,b}, gvp{wh.w, ws.w, ws.b, wv.w, wsv.w, wsv.b}},
//   We{...}, enc[3]{n0{g,b}, n1{g,b}, conv{m0,m1,m2}, ff{f0,f1}},
//   dec[3]{...}, Wout{wh.w, ws.w, ws.b}, dec_in{w,b},
//   time{w, l1{w,b}, l2{w,b}}                       (230 leaves, n_in == 238)
// ---------------------------------------------------------------------------

typedef __attribute__((ext_vector_type(16))) __bf16 v16bf;
typedef __attribute__((ext_vector_type(8)))  float  v8f;

// ---------------- model constants ----------------
#define KN_NODES 10000
#define KN_EDGES 120000
#define KCONF    2
#define KNI      (KN_NODES * KCONF)   // 20000 node instances (encoder)
#define KEI      (KN_EDGES * KCONF)   // 240000 edge instances (encoder)
#define KSIN 64
#define KVIN 4
#define KNS  128
#define KNV  16
#define KSE  32
#define KVE  1
#define KOUT 4
#define KMSG_SI (2*KNS + KSE)         // 288
#define KMSG_VI (2*KNV + KVE)         // 33

// ---------------- helpers ----------------
__device__ __forceinline__ __bf16 f2bf(float f) {
  unsigned u = __builtin_bit_cast(unsigned, f);
  unsigned r = u + 0x7FFFu + ((u >> 16) & 1u);      // round-to-nearest-even
  unsigned short h = (unsigned short)(r >> 16);
  return __builtin_bit_cast(__bf16, h);
}
__device__ __forceinline__ unsigned pack2bf(float lo, float hi) {
  unsigned a = (unsigned)__builtin_bit_cast(unsigned short, f2bf(lo));
  unsigned b = (unsigned)__builtin_bit_cast(unsigned short, f2bf(hi));
  return a | (b << 16);
}
__device__ __forceinline__ float sigmoidf_(float x) { return 1.f / (1.f + expf(-x)); }
__device__ __forceinline__ float waveRedSum(float v) {
#pragma unroll
  for (int o = 16; o > 0; o >>= 1) v += __shfl_xor(v, o, 32);
  return v;
}

// ---------------- gathered-segment descriptors ----------------
struct Seg  { const float* ptr; const int* idx; long long stride; int off; int width; };
struct GArgs { Seg seg[4]; int nseg; };
struct VSeg { const float* ptr; const int* idx; long long stride; int off; int nvec; };
struct VArgs { VSeg seg[3]; int nseg; };

// ---------------- generic WMMA GEMM:  Y[M x N] = gatherA[M x K] @ W[N x K]^T + b
#define GBM 64
#define GBN 64
#define GBK 64   // two 16x16x32 WMMA K-chunks per LDS tile

__global__ __launch_bounds__(256)
void gemm_gather(GArgs ga, const float* __restrict__ W, const float* __restrict__ bias,
                 float* __restrict__ Y, int M, int N, int K)
{
  __shared__ __bf16 As[GBM][GBK + 8];          // 64 x 72 bf16, rows 144B-aligned
  __shared__ __bf16 Bs[GBN][GBK + 8];
  __shared__ int rowOff[4][GBM];               // per-row 32-bit gather offsets

  const int tid  = threadIdx.x;
  const int lane = tid & 31;
  const int wid  = tid >> 5;
  const int wm   = wid & 3;        // 4 M-subtiles of 16
  const int wn   = wid >> 2;       // 2 N-subtiles of 32
  const int bm0  = blockIdx.y * GBM;
  const int bn0  = blockIdx.x * GBN;

  // Resolve per-row gather offsets once (kills idx loads + addr math in hot loop)
  {
    const int s = tid >> 6, r = tid & 63;      // 256 threads == 4*64 slots
    int ro = -1;
    if (s < ga.nseg) {
      const Seg sg = ga.seg[s];
      const int gr = bm0 + r;
      if (gr < M) {
        long long row = sg.idx ? (long long)sg.idx[gr] : (long long)gr;
        ro = (int)(row * sg.stride);           // fits in 32 bits for this model
      }
    }
    rowOff[s][r] = ro;
  }
  __syncthreads();

  // Each thread owns a fixed column pair {2cp, 2cp+1}; rows r = i*8 + r0.
  const int cp = tid & 31;
  const int r0 = tid >> 5;

  v8f acc0 = {}; v8f acc1 = {};
  const int nk = (K + GBK - 1) / GBK;

  for (int kt = 0; kt < nk; ++kt) {
    const int k0  = kt * GBK;
    const int gc0 = k0 + 2 * cp, gc1 = gc0 + 1;
    // Resolve segment for each of the two columns ONCE per K-tile.
    const float *p0 = nullptr, *p1 = nullptr;
    int s0 = -1, s1 = -1, o0 = 0, o1 = 0;
#pragma unroll
    for (int s = 0; s < 4; ++s) {
      if (s < ga.nseg) {
        const int o = ga.seg[s].off, w = ga.seg[s].width;
        if (gc0 < K && gc0 >= o && gc0 < o + w) { s0 = s; p0 = ga.seg[s].ptr; o0 = gc0 - o; }
        if (gc1 < K && gc1 >= o && gc1 < o + w) { s1 = s; p1 = ga.seg[s].ptr; o1 = gc1 - o; }
      }
    }
#pragma unroll
    for (int i = 0; i < 8; ++i) {
      const int r = i * 8 + r0;
      // A pair (gathered concat)
      float a0 = 0.f, a1 = 0.f;
      if (s0 >= 0) { const int ro = rowOff[s0][r]; if (ro >= 0) a0 = p0[ro + o0]; }
      if (s1 >= 0) { const int ro = rowOff[s1][r]; if (ro >= 0) a1 = p1[ro + o1]; }
      ((unsigned*)&As[r][0])[cp] = pack2bf(a0, a1);
      // B pair: W is (N x K) row-major
      const int gn = bn0 + r;
      float b0 = (gn < N && gc0 < K) ? W[(long long)gn * K + gc0] : 0.f;
      float b1 = (gn < N && gc1 < K) ? W[(long long)gn * K + gc1] : 0.f;
      ((unsigned*)&Bs[r][0])[cp] = pack2bf(b0, b1);
    }
    __syncthreads();

#pragma unroll
    for (int kk = 0; kk < GBK; kk += 32) {
      // A fragment: lane<16 -> row, K in {0..7,16..23}; lane>=16 -> {8..15,24..31}
      v16bf af, bf0, bf1;
      const int arow = wm * 16 + (lane & 15);
      const int kb   = (lane < 16) ? 0 : 8;
#pragma unroll
      for (int e2 = 0; e2 < 8; ++e2) {
        af[e2]     = As[arow][kk + kb + e2];
        af[e2 + 8] = As[arow][kk + 16 + kb + e2];
      }
      // B fragment: lane<16 -> col, K in {0..7,16..23}; lane>=16 -> {8..15,24..31}
      const int bcol = lane & 15;
      const int kb2  = (lane < 16) ? 0 : 8;
#pragma unroll
      for (int e2 = 0; e2 < 8; ++e2) {
        bf0[e2]     = Bs[wn * 32 + bcol][kk + kb2 + e2];
        bf0[e2 + 8] = Bs[wn * 32 + bcol][kk + 16 + kb2 + e2];
        bf1[e2]     = Bs[wn * 32 + 16 + bcol][kk + kb2 + e2];
        bf1[e2 + 8] = Bs[wn * 32 + 16 + bcol][kk + 16 + kb2 + e2];
      }
      acc0 = __builtin_amdgcn_wmma_f32_16x16x32_bf16(false, af, false, bf0, (short)0, acc0, false, false);
      acc1 = __builtin_amdgcn_wmma_f32_16x16x32_bf16(false, af, false, bf1, (short)0, acc1, false, false);
    }
    __syncthreads();
  }

  // C/D layout: element r of lane<16 -> (M=r, N=lane); lane>=16 -> (M=r+8, N=lane-16)
  const int ncol = lane & 15;
  const int mb   = (lane < 16) ? 0 : 8;
#pragma unroll
  for (int r = 0; r < 8; ++r) {
    const int gm = bm0 + wm * 16 + mb + r;
    if (gm >= M) continue;
    int gn = bn0 + wn * 32 + ncol;
    if (gn < N) Y[(long long)gm * N + gn] = acc0[r] + (bias ? bias[gn] : 0.f);
    gn += 16;
    if (gn < N) Y[(long long)gm * N + gn] = acc1[r] + (bias ? bias[gn] : 0.f);
  }
}

// ---------------- vector-channel kernels (3-vectors, tiny K -> VALU) --------
// One wave32 per row, 8 rows per block. All threads reach barriers.
// vh[r][j][c] = sum_k wh[j,k] * mv[r][k][c];  vhn[r][j] = sqrt(max(|vh_j|^2,1e-8))
__global__ __launch_bounds__(256)
void vec_vh(VArgs va, const float* __restrict__ wh,
            float* __restrict__ vh, float* __restrict__ vhn, int M, int vi, int h)
{
  __shared__ float smv[8][100];
  __shared__ float svh[8][100];
  const int w = threadIdx.x >> 5, lane = threadIdx.x & 31;
  const int r = blockIdx.x * 8 + w;
  const bool act = (r < M);
  if (act) {
    for (int t = lane; t < vi * 3; t += 32) {
      const int j = t / 3, c = t - 3 * j;
      float val = 0.f;
#pragma unroll
      for (int s = 0; s < 3; ++s) {
        if (s < va.nseg) {
          const VSeg sg = va.seg[s];
          if (j >= sg.off && j < sg.off + sg.nvec) {
            long long row = sg.idx ? (long long)sg.idx[r] : (long long)r;
            val = sg.ptr[row * sg.stride + (long long)(j - sg.off) * 3 + c];
          }
        }
      }
      smv[w][t] = val;
    }
  }
  __syncthreads();
  if (act) {
    for (int t = lane; t < h * 3; t += 32) {
      const int j = t / 3, c = t - 3 * j;
      float a = 0.f;
      for (int k = 0; k < vi; ++k) a += wh[j * vi + k] * smv[w][k * 3 + c];
      svh[w][t] = a;
      vh[(long long)r * h * 3 + t] = a;
    }
  }
  __syncthreads();
  if (act) {
    for (int j = lane; j < h; j += 32) {
      float ss = svh[w][j*3]*svh[w][j*3] + svh[w][j*3+1]*svh[w][j*3+1] + svh[w][j*3+2]*svh[w][j*3+2];
      vhn[(long long)r * h + j] = sqrtf(fmaxf(ss, 1e-8f));
    }
  }
}

// vout[r][i][c] = (sum_j wv[i,j]*vh[r][j][c]) * sigmoid(gate[r][i])  (+ resid)
__global__ __launch_bounds__(256)
void vec_post(const float* __restrict__ vh, const float* __restrict__ wv,
              const float* __restrict__ gate, float* __restrict__ vout,
              int M, int h, int vo, const float* __restrict__ resid)
{
  __shared__ float svh[8][100];
  const int w = threadIdx.x >> 5, lane = threadIdx.x & 31;
  const int r = blockIdx.x * 8 + w;
  const bool act = (r < M);
  if (act)
    for (int t = lane; t < h * 3; t += 32) svh[w][t] = vh[(long long)r * h * 3 + t];
  __syncthreads();
  if (act) {
    for (int t = lane; t < vo * 3; t += 32) {
      const int i = t / 3, c = t - 3 * i;
      float a = 0.f;
      for (int j = 0; j < h; ++j) a += wv[i * h + j] * svh[w][j * 3 + c];
      float o = a * sigmoidf_(gate[(long long)r * vo + i]);
      if (resid) o += resid[(long long)r * vo * 3 + t];
      vout[(long long)r * vo * 3 + t] = o;
    }
  }
}

// ---------------- LayerNorm (_ln): one wave32 per row -----------------------
__global__ __launch_bounds__(256)
void ln_kernel(const float* __restrict__ s, const float* __restrict__ v,
               const float* __restrict__ g, const float* __restrict__ b,
               float* __restrict__ os, float* __restrict__ ov, int M, int ds, int nv)
{
  const int lane = threadIdx.x & 31;
  const int row  = blockIdx.x * 8 + (threadIdx.x >> 5);
  if (row >= M) return;
  const float* sr = s + (long long)row * ds;
  float* orow = os + (long long)row * ds;
  float sum = 0.f, sum2 = 0.f;
  for (int i = lane; i < ds; i += 32) { float x = sr[i]; sum += x; sum2 += x * x; }
  sum = waveRedSum(sum); sum2 = waveRedSum(sum2);
  const float mu  = sum / ds;
  const float inv = rsqrtf(sum2 / ds - mu * mu + 1e-5f);
  for (int i = lane; i < ds; i += 32) orow[i] = (sr[i] - mu) * inv * g[i] + b[i];
  const float* vr = v + (long long)row * nv * 3;
  float* ovr = ov + (long long)row * nv * 3;
  float vs = 0.f;
  for (int j = lane; j < nv; j += 32) {
    float n2 = vr[j*3]*vr[j*3] + vr[j*3+1]*vr[j*3+1] + vr[j*3+2]*vr[j*3+2];
    vs += fmaxf(n2, 1e-8f);
  }
  vs = waveRedSum(vs);
  const float vinv = rsqrtf(vs / nv);
  for (int t = lane; t < nv * 3; t += 32) ovr[t] = vr[t] * vinv;
}

// ---------------- elementwise / scatter / pooling ---------------------------
__global__ void silu_ip(float* __restrict__ x, long long n) {
  long long i = blockIdx.x * (long long)blockDim.x + threadIdx.x;
  const long long st = (long long)gridDim.x * blockDim.x;
  for (; i < n; i += st) { float v = x[i]; x[i] = v / (1.f + expf(-v)); }
}
__global__ void add_ip(float* __restrict__ x, const float* __restrict__ y, long long n) {
  long long i = blockIdx.x * (long long)blockDim.x + threadIdx.x;
  const long long st = (long long)gridDim.x * blockDim.x;
  for (; i < n; i += st) x[i] += y[i];
}
__global__ void scatter_add(const float* __restrict__ src, const int* __restrict__ row,
                            int feat, float* __restrict__ dst, long long M) {
  const long long n = M * feat;
  long long i = blockIdx.x * (long long)blockDim.x + threadIdx.x;
  const long long st = (long long)gridDim.x * blockDim.x;
  for (; i < n; i += st) {
    const long long r = i / feat; const int f = (int)(i - r * feat);
    atomicAdd(&dst[(long long)row[r] * feat + f], src[i]);
  }
}
__global__ void agg_finish(float* __restrict__ x, const float* __restrict__ agg,
                           const float* __restrict__ cnt, int C, int feat, long long nInst) {
  const long long n = nInst * feat;
  long long i = blockIdx.x * (long long)blockDim.x + threadIdx.x;
  const long long st = (long long)gridDim.x * blockDim.x;
  for (; i < n; i += st) {
    const long long inst = i / feat; const long long node = inst / C;
    x[i] += agg[i] / fmaxf(cnt[node], 1.f);
  }
}
__global__ void count_dst(const int* __restrict__ dst, float* __restrict__ cnt, int E) {
  int i = blockIdx.x * blockDim.x + threadIdx.x;
  if (i < E) atomicAdd(&cnt[dst[i]], 1.f);
}
__global__ void expand_idx(const int* __restrict__ src, const int* __restrict__ dst,
                           int* __restrict__ srcRow, int* __restrict__ dstRow, int E, int C) {
  int i = blockIdx.x * blockDim.x + threadIdx.x;
  if (i < E * C) { const int e = i / C, c = i - e * C;
    srcRow[i] = src[e] * C + c; dstRow[i] = dst[e] * C + c; }
}
__global__ void ntrue_kernel(const float* __restrict__ mask, float* __restrict__ nt, int N, int C) {
  int n = blockIdx.x * blockDim.x + threadIdx.x;
  if (n < N) { float s = 0.f; for (int c = 0; c < C; ++c) s += mask[n * C + c]; nt[n] = s; }
}
__global__ void pool_nodes(const float* __restrict__ x, const float* __restrict__ mask,
                           const float* __restrict__ nt, float* __restrict__ y,
                           int N, int C, int feat) {
  const long long n = (long long)N * feat;
  long long i = blockIdx.x * (long long)blockDim.x + threadIdx.x;
  const long long st = (long long)gridDim.x * blockDim.x;
  for (; i < n; i += st) {
    const long long node = i / feat; const int f = (int)(i - node * feat);
    float a = 0.f;
    for (int c = 0; c < C; ++c) a += x[((long long)node * C + c) * feat + f] * mask[node * C + c];
    y[i] = a / nt[node];
  }
}
__global__ void pool_edges(const float* __restrict__ x, const float* __restrict__ mask,
                           const float* __restrict__ nt, const int* __restrict__ srcNode,
                           float* __restrict__ y, int E, int C, int feat) {
  const long long n = (long long)E * feat;
  long long i = blockIdx.x * (long long)blockDim.x + threadIdx.x;
  const long long st = (long long)gridDim.x * blockDim.x;
  for (; i < n; i += st) {
    const long long e = i / feat; const int f = (int)(i - e * feat);
    const int node = srcNode[e];
    float a = 0.f;
    for (int c = 0; c < C; ++c) a += x[((long long)e * C + c) * feat + f] * mask[node * C + c];
    y[i] = a / nt[node];
  }
}
// time embedding MLP: 17 -> 256 (silu) -> 128, single block
__global__ __launch_bounds__(256)
void time_mlp(const float* __restrict__ noise, const float* __restrict__ w8,
              const float* __restrict__ W1, const float* __restrict__ b1,
              const float* __restrict__ W2, const float* __restrict__ b2,
              float* __restrict__ t) {
  __shared__ float emb[17]; __shared__ float h1[256];
  const float x = noise[0];
  const int tid = threadIdx.x;
  if (tid == 0) emb[0] = x;
  if (tid < 8) { const float f = x * w8[tid] * 6.283185307179586f;
    emb[1 + tid] = sinf(f); emb[9 + tid] = cosf(f); }
  __syncthreads();
  { float a = b1[tid]; for (int k = 0; k < 17; ++k) a += W1[tid * 17 + k] * emb[k];
    h1[tid] = a / (1.f + expf(-a)); }
  __syncthreads();
  if (tid < 128) { float a = b2[tid]; for (int k = 0; k < 256; ++k) a += W2[tid * 256 + k] * h1[k];
    t[tid] = a; }
}

// ---------------------------------------------------------------------------
// Host-side orchestration
// ---------------------------------------------------------------------------
namespace {

struct LinP { const float* w; const float* b; };
struct GvpP { const float* wh; LinP ws; const float* wv; LinP wsv; int si, vi, so, vo, h; };
struct LnP  { const float* g; const float* b; };
struct LayerP { LnP n0, n1; GvpP m0, m1, m2, f0, f1; };

struct Cur { void* const* in; int i;
  const float* f() { return (const float*)in[i++]; } };

LnP parseLn(Cur& c) { LnP p; p.g = c.f(); p.b = c.f(); return p; }
GvpP parseGvp(Cur& c, int si, int vi, int so, int vo) {
  GvpP p; p.si = si; p.vi = vi; p.so = so; p.vo = vo; p.h = vi > vo ? vi : vo;
  p.wh = c.f(); p.ws.w = c.f(); p.ws.b = c.f();
  if (vo) { p.wv = c.f(); p.wsv.w = c.f(); p.wsv.b = c.f(); }
  else { p.wv = nullptr; p.wsv.w = nullptr; p.wsv.b = nullptr; }
  return p;
}
LayerP parseLayer(Cur& c) {
  LayerP L;
  L.n0 = parseLn(c); L.n1 = parseLn(c);
  L.m0 = parseGvp(c, KMSG_SI, KMSG_VI, KNS, KNV);
  L.m1 = parseGvp(c, KNS, KNV, KNS, KNV);
  L.m2 = parseGvp(c, KNS, KNV, KNS, KNV);
  L.f0 = parseGvp(c, KNS, KNV, 4 * KNS, 2 * KNV);
  L.f1 = parseGvp(c, 4 * KNS, 2 * KNV, KNS, KNV);
  return L;
}

struct Bufs {
  float *sA, *vA, *esA, *evA;        // encoder node/edge running features
  float *sB, *vB, *esB, *evB;        // post-pool (decoder)
  float *cnt, *n_true, *t_vec;
  int   *idxSrc, *idxDst;
  float *ln_s, *ln_v;
  float *e_vh, *e_vhn, *e_s0, *e_s1, *e_gate, *e_v0, *e_v1;
  float *n_vh, *n_vhn, *n_s0, *n_s1, *n_gate, *n_v0;
  float *agg_s, *agg_v;
};

inline dim3 ewGrid(long long n) {
  long long b = (n + 255) / 256; if (b > 8192) b = 8192; if (b < 1) b = 1;
  return dim3((unsigned)b);
}
inline dim3 rowGrid(int M) { return dim3((unsigned)((M + 7) / 8)); }
inline void launch_gemm(hipStream_t st, const GArgs& ga, const float* W, const float* b,
                        float* Y, int M, int N, int K) {
  dim3 g((N + GBN - 1) / GBN, (M + GBM - 1) / GBM);
  gemm_gather<<<g, 256, 0, st>>>(ga, W, b, Y, M, N, K);
}

// Full GVP: vh/vhn -> ws GEMM (pre-act) -> wsv gate GEMM -> vec_post -> SiLU(s)
void run_gvp(hipStream_t st, const GvpP& P, GArgs sIn, const VArgs& vIn, int M,
             float* vh, float* vhn, float* sOut, float* gate, float* vOut,
             const float* vResid, bool act_silu) {
  vec_vh<<<rowGrid(M), 256, 0, st>>>(vIn, P.wh, vh, vhn, M, P.vi, P.h);
  sIn.seg[sIn.nseg] = Seg{vhn, nullptr, (long long)P.h, P.si, P.h};
  sIn.nseg++;
  launch_gemm(st, sIn, P.ws.w, P.ws.b, sOut, M, P.so, P.si + P.h);
  if (P.vo > 0) {
    GArgs gg{}; gg.nseg = 1;
    gg.seg[0] = Seg{sOut, nullptr, (long long)P.so, 0, P.so};
    launch_gemm(st, gg, P.wsv.w, P.wsv.b, gate, M, P.vo, P.so);  // gate uses pre-act s
    vec_post<<<rowGrid(M), 256, 0, st>>>(vh, P.wv, gate, vOut, M, P.h, P.vo, vResid);
  }
  if (act_silu) { long long n = (long long)M * P.so; silu_ip<<<ewGrid(n), 256, 0, st>>>(sOut, n); }
}

void run_conv(hipStream_t st, const GvpP& m0, const GvpP& m1, const GvpP& m2,
              const float* sIn, const float* vIn, const float* es, const float* ev,
              const int* srcRow, const int* dstRow, int Medge, int nNodes, int C,
              float* sAcc, float* vAcc, const float* cnt, Bufs& B) {
  GArgs s0{}; s0.nseg = 3;
  s0.seg[0] = Seg{sIn, srcRow, KNS, 0, KNS};
  s0.seg[1] = Seg{es,  nullptr, KSE, KNS, KSE};
  s0.seg[2] = Seg{sIn, dstRow, KNS, KNS + KSE, KNS};
  VArgs v0{}; v0.nseg = 3;
  v0.seg[0] = VSeg{vIn, srcRow, 3 * KNV, 0, KNV};
  v0.seg[1] = VSeg{ev,  nullptr, 3 * KVE, KNV, KVE};
  v0.seg[2] = VSeg{vIn, dstRow, 3 * KNV, KNV + KVE, KNV};
  run_gvp(st, m0, s0, v0, Medge, B.e_vh, B.e_vhn, B.e_s0, B.e_gate, B.e_v0, nullptr, true);

  GArgs s1{}; s1.nseg = 1; s1.seg[0] = Seg{B.e_s0, nullptr, KNS, 0, KNS};
  VArgs v1{}; v1.nseg = 1; v1.seg[0] = VSeg{B.e_v0, nullptr, 3 * KNV, 0, KNV};
  run_gvp(st, m1, s1, v1, Medge, B.e_vh, B.e_vhn, B.e_s1, B.e_gate, B.e_v1, nullptr, true);

  GArgs s2{}; s2.nseg = 1; s2.seg[0] = Seg{B.e_s1, nullptr, KNS, 0, KNS};
  VArgs v2{}; v2.nseg = 1; v2.seg[0] = VSeg{B.e_v1, nullptr, 3 * KNV, 0, KNV};
  run_gvp(st, m2, s2, v2, Medge, B.e_vh, B.e_vhn, B.e_s0, B.e_gate, B.e_v0, nullptr, false);

  const long long nInst = (long long)nNodes * C;
  hipMemsetAsync(B.agg_s, 0, nInst * KNS * sizeof(float), st);
  hipMemsetAsync(B.agg_v, 0, nInst * 3 * KNV * sizeof(float), st);
  scatter_add<<<ewGrid((long long)Medge * KNS), 256, 0, st>>>(B.e_s0, dstRow, KNS, B.agg_s, Medge);
  scatter_add<<<ewGrid((long long)Medge * 3 * KNV), 256, 0, st>>>(B.e_v0, dstRow, 3 * KNV, B.agg_v, Medge);
  agg_finish<<<ewGrid(nInst * KNS), 256, 0, st>>>(sAcc, B.agg_s, cnt, C, KNS, nInst);
  agg_finish<<<ewGrid(nInst * 3 * KNV), 256, 0, st>>>(vAcc, B.agg_v, cnt, C, 3 * KNV, nInst);
}

void run_ff(hipStream_t st, const GvpP& f0, const GvpP& f1,
            const float* sIn, const float* vIn, int M,
            float* sAcc, float* vAcc, Bufs& B) {
  GArgs sa{}; sa.nseg = 1; sa.seg[0] = Seg{sIn, nullptr, KNS, 0, KNS};
  VArgs va{}; va.nseg = 1; va.seg[0] = VSeg{vIn, nullptr, 3 * KNV, 0, KNV};
  run_gvp(st, f0, sa, va, M, B.n_vh, B.n_vhn, B.n_s0, B.n_gate, B.n_v0, nullptr, true);
  GArgs sb{}; sb.nseg = 1; sb.seg[0] = Seg{B.n_s0, nullptr, 4 * KNS, 0, 4 * KNS};
  VArgs vb{}; vb.nseg = 1; vb.seg[0] = VSeg{B.n_v0, nullptr, 6 * KNV, 0, 2 * KNV};
  run_gvp(st, f1, sb, vb, M, B.n_vh, B.n_vhn, B.n_s1, B.n_gate, vAcc, vAcc /*resid*/, false);
  add_ip<<<ewGrid((long long)M * KNS), 256, 0, st>>>(sAcc, B.n_s1, (long long)M * KNS);
}

} // namespace

extern "C" void kernel_launch(void* const* d_in, const int* in_sizes, int n_in,
                              void* d_out, int out_size, void* d_ws, size_t ws_size,
                              hipStream_t stream) {
  (void)in_sizes; (void)n_in; (void)out_size; (void)ws_size;
  const float* node_s = (const float*)d_in[0];
  const float* node_v = (const float*)d_in[1];
  const float* edge_s = (const float*)d_in[2];
  const float* edge_v = (const float*)d_in[3];
  const int*   eidx   = (const int*)d_in[4];
  const float* mask   = (const float*)d_in[5];
  const float* z_t    = (const float*)d_in[6];
  const float* noise  = (const float*)d_in[7];
  const int* src = eidx;
  const int* dst = eidx + KN_EDGES;

  // ---- parse params (insertion-order recursive flatten; see header note) ----
  Cur c{d_in, 8};
  LnP  WvLn = parseLn(c);
  GvpP WvG  = parseGvp(c, KSIN, KVIN, KNS, KNV);
  LnP  WeLn = parseLn(c);
  GvpP WeG  = parseGvp(c, KSE, KVE, KSE, KVE);
  LayerP enc[3]; for (int l = 0; l < 3; ++l) enc[l] = parseLayer(c);
  LayerP dec[3]; for (int l = 0; l < 3; ++l) dec[l] = parseLayer(c);
  GvpP WoutG = parseGvp(c, KNS, KNV, KOUT, 0);
  LinP decIn{c.f(), c.f()};
  const float* timeW = c.f();
  LinP tl1{c.f(), c.f()};
  LinP tl2{c.f(), c.f()};

  // ---- workspace layout (deterministic bump allocation) ----
  char* wp = (char*)d_ws;
  auto allocF = [&](long long n) -> float* {
    float* p = (float*)wp; wp += ((n * 4 + 255) / 256) * 256; return p; };
  Bufs B;
  B.sA   = allocF((long long)KNI * KNS);
  B.vA   = allocF((long long)KNI * 3 * KNV);
  B.esA  = allocF((long long)KEI * KSE);
  B.evA  = allocF((long long)KEI * 3 * KVE);
  B.sB   = allocF((long long)KN_NODES * KNS);
  B.vB   = allocF((long long)KN_NODES * 3 * KNV);
  B.esB  = allocF((long long)KN_EDGES * KSE);
  B.evB  = allocF((long long)KN_EDGES * 3 * KVE);
  B.cnt  = allocF(KN_NODES);
  B.n_true = allocF(KN_NODES);
  B.t_vec  = allocF(KNS);
  B.idxSrc = (int*)allocF(KEI);
  B.idxDst = (int*)allocF(KEI);
  B.ln_s = allocF((long long)KNI * KNS);
  B.ln_v = allocF((long long)KNI * 3 * KNV);
  B.e_vh  = allocF((long long)KEI * KMSG_VI * 3);
  B.e_vhn = allocF((long long)KEI * KMSG_VI);
  B.e_s0  = allocF((long long)KEI * KNS);
  B.e_s1  = allocF((long long)KEI * KNS);
  B.e_gate= allocF((long long)KEI * KNV);
  B.e_v0  = allocF((long long)KEI * 3 * KNV);
  B.e_v1  = allocF((long long)KEI * 3 * KNV);
  B.n_vh  = allocF((long long)KNI * 3 * 2 * KNV);
  B.n_vhn = allocF((long long)KNI * 2 * KNV);
  B.n_s0  = allocF((long long)KNI * 4 * KNS);
  B.n_s1  = allocF((long long)KNI * KNS);
  B.n_gate= allocF((long long)KNI * 2 * KNV);
  B.n_v0  = allocF((long long)KNI * 3 * 2 * KNV);
  B.agg_s = allocF((long long)KNI * KNS);
  B.agg_v = allocF((long long)KNI * 3 * KNV);

  // ---- graph structure prep (recomputed each call; deterministic) ----
  hipMemsetAsync(B.cnt, 0, KN_NODES * sizeof(float), stream);
  count_dst<<<(KN_EDGES + 255) / 256, 256, 0, stream>>>(dst, B.cnt, KN_EDGES);
  expand_idx<<<(KEI + 255) / 256, 256, 0, stream>>>(src, dst, B.idxSrc, B.idxDst, KN_EDGES, KCONF);

  // ---- input embeddings ----
  ln_kernel<<<rowGrid(KNI), 256, 0, stream>>>(node_s, node_v, WvLn.g, WvLn.b,
                                              B.ln_s, B.ln_v, KNI, KSIN, KVIN);
  {
    GArgs sa{}; sa.nseg = 1; sa.seg[0] = Seg{B.ln_s, nullptr, KSIN, 0, KSIN};
    VArgs va{}; va.nseg = 1; va.seg[0] = VSeg{B.ln_v, nullptr, 3 * KVIN, 0, KVIN};
    run_gvp(stream, WvG, sa, va, KNI, B.n_vh, B.n_vhn, B.sA, B.n_gate, B.vA, nullptr, false);
  }
  ln_kernel<<<rowGrid(KEI), 256, 0, stream>>>(edge_s, edge_v, WeLn.g, WeLn.b,
                                              B.e_s1, B.e_v1, KEI, KSE, KVE);
  {
    GArgs sa{}; sa.nseg = 1; sa.seg[0] = Seg{B.e_s1, nullptr, KSE, 0, KSE};
    VArgs va{}; va.nseg = 1; va.seg[0] = VSeg{B.e_v1, nullptr, 3 * KVE, 0, KVE};
    run_gvp(stream, WeG, sa, va, KEI, B.e_vh, B.e_vhn, B.esA, B.e_gate, B.evA, nullptr, false);
  }

  // ---- encoder layers ----
  for (int l = 0; l < 3; ++l) {
    const LayerP& L = enc[l];
    ln_kernel<<<rowGrid(KNI), 256, 0, stream>>>(B.sA, B.vA, L.n0.g, L.n0.b,
                                                B.ln_s, B.ln_v, KNI, KNS, KNV);
    run_conv(stream, L.m0, L.m1, L.m2, B.ln_s, B.ln_v, B.esA, B.evA,
             B.idxSrc, B.idxDst, KEI, KN_NODES, KCONF, B.sA, B.vA, B.cnt, B);
    ln_kernel<<<rowGrid(KNI), 256, 0, stream>>>(B.sA, B.vA, L.n1.g, L.n1.b,
                                                B.ln_s, B.ln_v, KNI, KNS, KNV);
    run_ff(stream, L.f0, L.f1, B.ln_s, B.ln_v, KNI, B.sA, B.vA, B);
  }

  // ---- conformer pooling (masked mean over NCONF) ----
  ntrue_kernel<<<(KN_NODES + 255) / 256, 256, 0, stream>>>(mask, B.n_true, KN_NODES, KCONF);
  pool_nodes<<<ewGrid((long long)KN_NODES * KNS), 256, 0, stream>>>(B.sA, mask, B.n_true, B.sB, KN_NODES, KCONF, KNS);
  pool_nodes<<<ewGrid((long long)KN_NODES * 3 * KNV), 256, 0, stream>>>(B.vA, mask, B.n_true, B.vB, KN_NODES, KCONF, 3 * KNV);
  pool_edges<<<ewGrid((long long)KN_EDGES * KSE), 256, 0, stream>>>(B.esA, mask, B.n_true, src, B.esB, KN_EDGES, KCONF, KSE);
  pool_edges<<<ewGrid((long long)KN_EDGES * 3 * KVE), 256, 0, stream>>>(B.evA, mask, B.n_true, src, B.evB, KN_EDGES, KCONF, 3 * KVE);

  // ---- time embedding + dec_in ----
  time_mlp<<<1, 256, 0, stream>>>(noise, timeW, tl1.w, tl1.b, tl2.w, tl2.b, B.t_vec);
  {
    GArgs ga{}; ga.nseg = 2;
    ga.seg[0] = Seg{z_t,     nullptr, KOUT, 0, KOUT};
    ga.seg[1] = Seg{B.t_vec, nullptr, 0,    KOUT, KNS};   // rowStride 0 => broadcast
    launch_gemm(stream, ga, decIn.w, decIn.b, B.n_s1, KN_NODES, KNS, KOUT + KNS);
    add_ip<<<ewGrid((long long)KN_NODES * KNS), 256, 0, stream>>>(B.sB, B.n_s1, (long long)KN_NODES * KNS);
  }

  // ---- decoder layers (conv -> LN -> ff -> LN) ----
  for (int l = 0; l < 3; ++l) {
    const LayerP& L = dec[l];
    run_conv(stream, L.m0, L.m1, L.m2, B.sB, B.vB, B.esB, B.evB,
             src, dst, KN_EDGES, KN_NODES, 1, B.sB, B.vB, B.cnt, B);
    ln_kernel<<<rowGrid(KN_NODES), 256, 0, stream>>>(B.sB, B.vB, L.n0.g, L.n0.b,
                                                     B.sB, B.vB, KN_NODES, KNS, KNV);
    run_ff(stream, L.f0, L.f1, B.sB, B.vB, KN_NODES, B.sB, B.vB, B);
    ln_kernel<<<rowGrid(KN_NODES), 256, 0, stream>>>(B.sB, B.vB, L.n1.g, L.n1.b,
                                                     B.sB, B.vB, KN_NODES, KNS, KNV);
  }

  // ---- output head (Wout: GVP with vo=0 -> scalars only) ----
  {
    VArgs va{}; va.nseg = 1; va.seg[0] = VSeg{B.vB, nullptr, 3 * KNV, 0, KNV};
    vec_vh<<<rowGrid(KN_NODES), 256, 0, stream>>>(va, WoutG.wh, B.n_vh, B.n_vhn, KN_NODES, KNV, KNV);
    GArgs ga{}; ga.nseg = 2;
    ga.seg[0] = Seg{B.sB,    nullptr, KNS, 0, KNS};
    ga.seg[1] = Seg{B.n_vhn, nullptr, KNV, KNS, KNV};
    launch_gemm(stream, ga, WoutG.ws.w, WoutG.ws.b, (float*)d_out, KN_NODES, KOUT, KNS + KNV);
  }
}
